// LSTM_86440511799910
// MI455X (gfx1250) — matrix operand order repaired and checked by reading, not compiled, not run on hardware
//
#include <hip/hip_runtime.h>
#include <math.h>

#define NWG   16
#define HDIM  256
#define BSZ   256
#define TLEN  784

// smem layout (bytes)
#define OFF_HT     0         // [B][H] bf16           131072
#define OFF_WLDS   131072    // 4 gates x 8k x 32 x16  32768
#define OFF_GATES  163840    // [64][256] bf16         32768
#define OFF_CST    196608    // [16][256] f32          16384
#define OFF_XS     212992    // [256] f32               1024
#define OFF_WXL    214016    // [64] f32                 256
#define OFF_BL     214272    // [64] f32                 256
#define SMEMB      214528

// CDNA5 async global->LDS path (ASYNCcnt), guarded for toolchain portability
#if defined(__has_builtin)
#  if __has_builtin(__builtin_amdgcn_global_load_async_to_lds_b128) && \
      __has_builtin(__builtin_amdgcn_s_wait_asynccnt)
#    define USE_ASYNC_LDS 1
#  endif
#endif
#ifndef USE_ASYNC_LDS
#  define USE_ASYNC_LDS 0
#endif

typedef __attribute__((ext_vector_type(16))) __bf16 v16bf;
typedef __attribute__((ext_vector_type(8)))  float  v8f;

#if USE_ASYNC_LDS
typedef int v4i_ __attribute__((vector_size(16)));
typedef v4i_ __attribute__((address_space(1)))* v4i_gptr;   // global
typedef v4i_ __attribute__((address_space(3)))* v4i_lptr;   // LDS
#endif

__device__ __forceinline__ float sigmoidf_(float z) { return 1.0f / (1.0f + __expf(-z)); }

// -------- prep: swizzle 4 recurrent weights into WMMA A-fragment layout (bf16),
// zero h buffer 0 and the grid-barrier counter. Idempotent per launch. --------
__global__ void __launch_bounds__(256)
lstm_prep_kernel(const float* __restrict__ Wgh, const float* __restrict__ Wih,
                 const float* __restrict__ Wfh, const float* __restrict__ Woh,
                 __bf16* __restrict__ Wfrag, __bf16* __restrict__ hbuf0,
                 unsigned* __restrict__ counter)
{
  int idx = blockIdx.x * 256 + threadIdx.x;     // 0..262143 : [mtile 64][k 8][lane 32][half 16]
  int m  = idx >> 12;
  int rm = idx & 4095;
  int k  = rm >> 9;
  int r2 = rm & 511;
  int l  = r2 >> 4;
  int p  = r2 & 15;
  // A-matrix 16x32 bf16 fragment: M = lane%16; K = 16*(lane/16) + element (contiguous, same as B)
  int Mg    = m * 16 + (l & 15);                // global Z row
  int khalf = l >> 4;
  int j     = p >> 1;
  int s     = p & 1;
  int K     = khalf * 16 + p;
  int col   = k * 32 + K;
  int gate  = Mg >> 8;
  int r     = Mg & 255;
  const float* Wsel = (gate == 0) ? Wgh : (gate == 1) ? Wih : (gate == 2) ? Wfh : Woh;
  Wfrag[idx] = (__bf16)Wsel[r * HDIM + col];

  if (idx < HDIM * BSZ) hbuf0[idx] = (__bf16)0.0f;
  if (idx == 0) *counter = 0u;
}

// -------- persistent LSTM: 16 WGs x 256 threads (8 waves). WG wg owns h rows
// [wg*16, wg*16+16) for all 4 gates. One grid barrier per timestep. --------
__global__ void __launch_bounds__(256, 1)
lstm_main_kernel(const float* __restrict__ x,
                 const float* __restrict__ Wgx, const float* __restrict__ bg,
                 const float* __restrict__ Wix, const float* __restrict__ bi,
                 const float* __restrict__ Wfx, const float* __restrict__ bf_,
                 const float* __restrict__ Wox, const float* __restrict__ bo,
                 const float* __restrict__ Wph, const float* __restrict__ bp,
                 const __bf16* __restrict__ Wfrag,
                 __bf16* __restrict__ hbuf0, __bf16* __restrict__ hbuf1,
                 unsigned* __restrict__ counter,
                 float* __restrict__ out)
{
  extern __shared__ __align__(128) char smem[];
  __bf16* hT     = (__bf16*)(smem + OFF_HT);      // h, batch-major [b][hcol]
  __bf16* Wlds   = (__bf16*)(smem + OFF_WLDS);    // A fragments [gate][k][lane][16]
  __bf16* gatesL = (__bf16*)(smem + OFF_GATES);   // activated gates [gate][16][256]
  float*  cst    = (float*) (smem + OFF_CST);     // c state [16][256]
  float*  xs     = (float*) (smem + OFF_XS);      // x_t [256]
  float*  wx_l   = (float*) (smem + OFF_WXL);     // per-WG x-weights [64]
  float*  b_l    = (float*) (smem + OFF_BL);      // per-WG biases    [64]

  const int tid   = threadIdx.x;
  const int wg    = blockIdx.x;
  const int wave  = tid >> 5;
  const int lane  = tid & 31;
  const int gate  = wave >> 1;          // wave pair per gate
  const int nbase = (wave & 1) * 8;     // batch-tile half

  // per-WG x-weights & biases
  if (tid < 64) {
    int g  = tid >> 4;
    int rr = wg * 16 + (tid & 15);
    const float* wxp = (g == 0) ? Wgx : (g == 1) ? Wix : (g == 2) ? Wfx : Wox;
    const float* bpp = (g == 0) ? bg  : (g == 1) ? bi  : (g == 2) ? bf_ : bo;
    wx_l[tid] = wxp[rr];
    b_l[tid]  = bpp[rr];
  }
  // zero c state
  #pragma unroll
  for (int i = 0; i < 16; i++) cst[tid + i * 256] = 0.0f;

  // cache this WG's A fragments in LDS (reused 784x)
  {
    const uint4* wsrc = (const uint4*)Wfrag;
    uint4*       wdst = (uint4*)Wlds;
    int ks = (wave & 1) * 4;
    for (int kk = 0; kk < 4; kk++) {
      int k  = ks + kk;
      int gi = (((gate * 16 + wg) * 8 + k) * 32 + lane) * 2;
      int li = ((gate * 8 + k) * 32 + lane) * 2;
      wdst[li]     = wsrc[gi];
      wdst[li + 1] = wsrc[gi + 1];
    }
  }

  unsigned barTarget = 0;

  for (int t = 0; t < TLEN; t++) {
    const __bf16* hcur  = (t & 1) ? hbuf1 : hbuf0;
    __bf16*       hnext = (t & 1) ? hbuf0 : hbuf1;

    // stage full h (B x H, bf16) into LDS
    {
      const uint4* src = (const uint4*)hcur;
      uint4*       dst = (uint4*)hT;
#if USE_ASYNC_LDS
      #pragma unroll
      for (int i = 0; i < 32; i++) {
        __builtin_amdgcn_global_load_async_to_lds_b128(
            (v4i_gptr)(src + tid + i * 256),
            (v4i_lptr)(dst + tid + i * 256),
            0, 0);
      }
      __builtin_amdgcn_s_wait_asynccnt(0);
#else
      #pragma unroll
      for (int i = 0; i < 32; i++) dst[tid + i * 256] = src[tid + i * 256];
#endif
    }
    xs[tid] = x[tid * TLEN + t];
    __syncthreads();

    // GEMM: wave computes 16 rows (its gate slice) x 128 batch cols
    v8f acc[8];
    #pragma unroll
    for (int ni = 0; ni < 8; ni++)
      #pragma unroll
      for (int e = 0; e < 8; e++) acc[ni][e] = 0.0f;

    #pragma unroll
    for (int k = 0; k < 8; k++) {
      v16bf a = *(const v16bf*)(Wlds + ((gate * 8 + k) * 32 + lane) * 16);
      const int kb = k * 32 + (lane >> 4) * 16;   // h column base (K)
      const int bl = lane & 15;                   // N within tile
      #pragma unroll
      for (int ni = 0; ni < 8; ni++) {
        int n = nbase + ni;
        v16bf bfr = *(const v16bf*)(hT + ((n * 16 + bl) * HDIM + kb));
        acc[ni] = __builtin_amdgcn_wmma_f32_16x16x32_bf16(
            false, a, false, bfr, (short)0, acc[ni], false, false);
      }
    }

    // epilogue: + Wx*x_t + bias, activation, stash in LDS
    const int mhi = 8 * (lane >> 4);
    #pragma unroll
    for (int ni = 0; ni < 8; ni++) {
      int bcol = (nbase + ni) * 16 + (lane & 15);
      float xv = xs[bcol];
      #pragma unroll
      for (int r = 0; r < 8; r++) {
        int hrl = r + mhi;                 // 0..15 (C layout: M = r + 8*(lane>>4))
        int zl  = gate * 16 + hrl;         // 0..63
        float z  = acc[ni][r] + wx_l[zl] * xv + b_l[zl];
        float av = (gate == 0) ? tanhf(z) : sigmoidf_(z);
        gatesL[zl * 256 + bcol] = (__bf16)av;
      }
    }
    __syncthreads();

    // c/h update (WG-local), write own h slice batch-major to global
    {
      int hr = tid >> 4;                   // 0..15
      int bb = (tid & 15) * 16;
      #pragma unroll
      for (int j2 = 0; j2 < 16; j2++) {
        int b = bb + j2;
        float g = (float)gatesL[(0 * 16 + hr) * 256 + b];
        float i = (float)gatesL[(1 * 16 + hr) * 256 + b];
        float f = (float)gatesL[(2 * 16 + hr) * 256 + b];
        float o = (float)gatesL[(3 * 16 + hr) * 256 + b];
        float cv = cst[hr * 256 + b];
        cv = g * i + cv * f;
        cst[hr * 256 + b] = cv;
        hnext[b * HDIM + wg * 16 + hr] = (__bf16)(tanhf(cv) * o);
      }
    }

    // grid barrier (counter monotonically increases; re-zeroed by prep kernel)
    barTarget += NWG;
    __threadfence();
    __syncthreads();
    if (tid == 0) {
      atomicAdd(counter, 1u);
      while (atomicAdd(counter, 0u) < barTarget) __builtin_amdgcn_s_sleep(1);
      __threadfence();
    }
    __syncthreads();
  }

  // projection: out[b][c] = sum_k Wph[c][k] * h[k][b] + bp[c]   (final h in hbuf0)
  if (wg == 0) {
    const __bf16* hfin = hbuf0;
    int b = tid;
    #pragma unroll
    for (int c = 0; c < 10; c++) {
      float s = bp[c];
      for (int k2 = 0; k2 < HDIM; k2++)
        s += Wph[c * HDIM + k2] * (float)hfin[b * HDIM + k2];
      out[b * 10 + c] = s;
    }
  }
}

extern "C" void kernel_launch(void* const* d_in, const int* in_sizes, int n_in,
                              void* d_out, int out_size, void* d_ws, size_t ws_size,
                              hipStream_t stream) {
  const float* x   = (const float*)d_in[0];
  const float* Wgx = (const float*)d_in[1];
  const float* Wgh = (const float*)d_in[2];
  const float* bg  = (const float*)d_in[3];
  const float* Wix = (const float*)d_in[4];
  const float* Wih = (const float*)d_in[5];
  const float* bi  = (const float*)d_in[6];
  const float* Wfx = (const float*)d_in[7];
  const float* Wfh = (const float*)d_in[8];
  const float* bf_ = (const float*)d_in[9];
  const float* Wox = (const float*)d_in[10];
  const float* Woh = (const float*)d_in[11];
  const float* bo  = (const float*)d_in[12];
  const float* Wph = (const float*)d_in[13];
  const float* bp  = (const float*)d_in[14];
  float* out = (float*)d_out;

  char* ws = (char*)d_ws;
  unsigned* counter = (unsigned*)ws;                         //    4 B
  __bf16* hbuf0 = (__bf16*)(ws + 1024);                      // 128 KB
  __bf16* hbuf1 = (__bf16*)(ws + 1024 + 131072);             // 128 KB
  __bf16* Wfrag = (__bf16*)(ws + 1024 + 262144);             // 512 KB

  (void)in_sizes; (void)n_in; (void)out_size; (void)ws_size;

  (void)hipFuncSetAttribute((const void*)lstm_main_kernel,
                            hipFuncAttributeMaxDynamicSharedMemorySize, SMEMB);

  lstm_prep_kernel<<<1024, 256, 0, stream>>>(Wgh, Wih, Wfh, Woh, Wfrag, hbuf0, counter);
  lstm_main_kernel<<<NWG, 256, SMEMB, stream>>>(
      x, Wgx, bg, Wix, bi, Wfx, bf_, Wox, bo, Wph, bp,
      Wfrag, hbuf0, hbuf1, counter, out);
}
